// VectorQuantizer_15891378995320
// MI455X (gfx1250) — matrix operand order repaired and checked by reading, not compiled, not run on hardware
//
#include <hip/hip_runtime.h>

#define N_EMB 4096
#define EMB_DIM 256
#define N_VEC 32768                     // 32*32*32 rows of z
#define ROWS_PER_WAVE 32                // two 16-row A operands per wave
#define WAVES_PER_BLOCK 4
#define ROWS_PER_BLOCK (ROWS_PER_WAVE * WAVES_PER_BLOCK)   // 128
#define NUM_BLOCKS (N_VEC / ROWS_PER_BLOCK)                // 256
#define NUM_WAVES (NUM_BLOCKS * WAVES_PER_BLOCK)           // 1024
#define EMB_SCALE 2048.0f               // lift uniform(+-1/4096) into f16 normal range

typedef _Float16 v16h __attribute__((ext_vector_type(16)));
typedef _Float16 v8h  __attribute__((ext_vector_type(8)));
typedef float    v8f  __attribute__((ext_vector_type(8)));
typedef float    v4f  __attribute__((ext_vector_type(4)));

// ---------------------------------------------------------------------------
// Phase 1: codebook f32 -> scaled f16 (WMMA B operand) + exact f32 norms.
// ---------------------------------------------------------------------------
__global__ __launch_bounds__(32) void vq_prep(const float* __restrict__ emb,
                                              _Float16* __restrict__ eh,
                                              float* __restrict__ norms) {
    const int row  = blockIdx.x;
    const int lane = threadIdx.x;
    const float* p = emb + (size_t)row * EMB_DIM + lane * 8;
    float s = 0.0f;
    v8h h;
#pragma unroll
    for (int i = 0; i < 8; ++i) {
        float v = p[i];
        s += v * v;
        h[i] = (_Float16)(v * EMB_SCALE);
    }
    *(v8h*)(eh + (size_t)row * EMB_DIM + lane * 8) = h;
#pragma unroll
    for (int m = 16; m >= 1; m >>= 1) s += __shfl_xor(s, m, 32);
    if (lane == 0) norms[row] = s;
}

// ---------------------------------------------------------------------------
// Phase 2: fused (-2 z.E^T + ||e||^2) GEMM + argmin + gather + ST output.
// Wave owns 32 z rows (two register-resident A operands). B chunks stream
// through a 4-deep rotating register buffer with 2-chunk lookahead. A-staging
// is fenced with sched_barriers so its one-time load/convert burst does not
// inflate the static VGPR allocation past 256 (avoids VGPR-MSB mode, keeps
// occupancy headroom for a second wave per SIMD).
// ---------------------------------------------------------------------------
__global__ __launch_bounds__(128) void vq_main(const float* __restrict__ z,
                                               const float* __restrict__ emb,
                                               const _Float16* __restrict__ eh,
                                               const float* __restrict__ norms,
                                               float* __restrict__ out,
                                               float* __restrict__ partials) {
    const int lane    = threadIdx.x & 31;
    const int wave    = threadIdx.x >> 5;
    const int rowBase = blockIdx.x * ROWS_PER_BLOCK + wave * ROWS_PER_WAVE;
    const int m  = lane & 15;   // A row within tile / B entry within tile
    const int hi = lane >> 4;   // lane half (ISA K-split for 16-bit A/B)

    // ---- Two A operands: rows [rowBase, +16) and [rowBase+16, +32).
    // Lane (m,hi) holds row m, K = kc*32 + hi*16 + {0..15}.
    v16h a0[8], a1[8];
    __builtin_amdgcn_sched_barrier(0);
#pragma unroll
    for (int t = 0; t < 2; ++t) {
        const float* zr = z + (size_t)(rowBase + t * 16 + m) * EMB_DIM;
#pragma unroll
        for (int kc = 0; kc < 8; ++kc) {
            const float* p = zr + kc * 32 + hi * 16;
            v16h av;
#pragma unroll
            for (int i = 0; i < 8; ++i) av[i]     = (_Float16)p[i];
#pragma unroll
            for (int i = 0; i < 8; ++i) av[8 + i] = (_Float16)p[8 + i];
            if (t == 0) a0[kc] = av; else a1[kc] = av;
            // Fence: keep only ~one chunk of f32 loads + cvt temps in flight,
            // capping peak VGPR pressure in the prologue.
            __builtin_amdgcn_sched_barrier(0);
        }
    }
    __builtin_amdgcn_sched_barrier(0);

    float minval0[8], minval1[8];
    int   minidx0[8], minidx1[8];
#pragma unroll
    for (int r = 0; r < 8; ++r) {
        minval0[r] = 3.0e38f; minidx0[r] = 0;
        minval1[r] = 3.0e38f; minidx1[r] = 0;
    }

    const float c2 = 2.0f / EMB_SCALE;   // undo EMB_SCALE inside -2*z.e

    // 4-deep rotating B buffer; preload chunks 0,1 of tile 0.
    const _Float16* base0 = eh + (size_t)m * EMB_DIM + hi * 16;
    v16h b[4];
    b[0] = *(const v16h*)(base0 + 0 * 32);
    b[1] = *(const v16h*)(base0 + 1 * 32);

    for (int nb = 0; nb < N_EMB; nb += 16) {
        const _Float16* bbase = eh + (size_t)(nb + m) * EMB_DIM + hi * 16;
        // Branchless next-tile base (clamped on the last tile: harmless reload).
        const _Float16* nbase = (nb + 16 < N_EMB) ? (bbase + 16 * EMB_DIM) : bbase;
        // Warm WGP$ two tiles ahead.
        if (nb + 32 < N_EMB)
            __builtin_prefetch(eh + (size_t)(nb + 32 + m) * EMB_DIM + hi * 16, 0, 1);

        v8f acc0 = {}, acc1 = {};
#pragma unroll
        for (int kc = 0; kc < 8; ++kc) {
            // Lookahead: issue the load of chunk kc+2 (possibly next tile's
            // chunk kc-6) into a buffer slot distinct from the two pending.
            if (kc < 6) b[(kc + 2) & 3] = *(const v16h*)(bbase + (kc + 2) * 32);
            else        b[(kc + 2) & 3] = *(const v16h*)(nbase + (kc - 6) * 32);

            acc0 = __builtin_amdgcn_wmma_f32_16x16x32_f16(
                       false, a0[kc], false, b[kc & 3], (short)0, acc0, false, false);
            acc1 = __builtin_amdgcn_wmma_f32_16x16x32_f16(
                       false, a1[kc], false, b[kc & 3], (short)0, acc1, false, false);
        }

        // C layout: this lane holds column n = nb+m for rows (hi*8 + r).
        const float nrm = norms[nb + m];
        const int   e   = nb + m;
#pragma unroll
        for (int r = 0; r < 8; ++r) {
            float d0 = __builtin_fmaf(-c2, acc0[r], nrm);  // ||e||^2 - 2 z.e
            float d1 = __builtin_fmaf(-c2, acc1[r], nrm);
            if (d0 < minval0[r]) { minval0[r] = d0; minidx0[r] = e; }
            if (d1 < minval1[r]) { minval1[r] = d1; minidx1[r] = e; }
        }
    }

    // Butterfly argmin across the 16 lanes of each half.
    // Tie-break on smaller codebook index == jnp.argmin first-occurrence.
#pragma unroll
    for (int r = 0; r < 8; ++r) {
        float mv0 = minval0[r]; int mi0 = minidx0[r];
        float mv1 = minval1[r]; int mi1 = minidx1[r];
#pragma unroll
        for (int msk = 1; msk < 16; msk <<= 1) {
            float ov0 = __shfl_xor(mv0, msk, 32); int oi0 = __shfl_xor(mi0, msk, 32);
            float ov1 = __shfl_xor(mv1, msk, 32); int oi1 = __shfl_xor(mi1, msk, 32);
            if (ov0 < mv0 || (ov0 == mv0 && oi0 < mi0)) { mv0 = ov0; mi0 = oi0; }
            if (ov1 < mv1 || (ov1 == mv1 && oi1 < mi1)) { mv1 = ov1; mi1 = oi1; }
        }
        minidx0[r] = mi0; minidx1[r] = mi1;
    }

    // Gather original-f32 codebook rows, straight-through output, loss.
    float lossAcc = 0.0f;
#pragma unroll
    for (int rr = 0; rr < 32; ++rr) {
        const int r   = rr & 7;
        const int src = ((rr >> 3) & 1) << 4;            // lane 0 or lane 16
        const int idx = (rr < 16) ? __shfl(minidx0[r], src, 32)
                                  : __shfl(minidx1[r], src, 32);
        const int row = rowBase + rr;
        const v4f* ev = (const v4f*)(emb + (size_t)idx * EMB_DIM);
        const v4f* zv = (const v4f*)(z   + (size_t)row * EMB_DIM);
        v4f*       ov = (v4f*)(out + (size_t)row * EMB_DIM);
#pragma unroll
        for (int t = 0; t < 2; ++t) {
            const int c = lane + t * 32;                 // 64 v4f per row
            v4f e4 = ev[c], z4 = zv[c];
            v4f d  = e4 - z4;                            // z_q - z
            v4f o4 = z4 + d;                             // z + (z_q - z), as in ref
            __builtin_nontemporal_store(o4, ov + c);     // stream, don't pollute L2
            lossAcc += d.x * d.x + d.y * d.y + d.z * d.z + d.w * d.w;
        }
    }
#pragma unroll
    for (int msk = 16; msk >= 1; msk >>= 1) lossAcc += __shfl_xor(lossAcc, msk, 32);
    if (lane == 0) partials[blockIdx.x * WAVES_PER_BLOCK + wave] = lossAcc;
}

// ---------------------------------------------------------------------------
// Phase 3: deterministic fixed-order reduction of wave partials.
// loss = (1 + 0.25) * mean((z_q - z)^2)
// ---------------------------------------------------------------------------
__global__ void vq_finalize(const float* __restrict__ partials,
                            float* __restrict__ out) {
    float s = 0.0f;
    for (int i = 0; i < NUM_WAVES; ++i) s += partials[i];
    out[(size_t)N_VEC * EMB_DIM] = 1.25f * s / (float)((size_t)N_VEC * EMB_DIM);
}

extern "C" void kernel_launch(void* const* d_in, const int* in_sizes, int n_in,
                              void* d_out, int out_size, void* d_ws, size_t ws_size,
                              hipStream_t stream) {
    (void)in_sizes; (void)n_in; (void)out_size; (void)ws_size;
    const float* z   = (const float*)d_in[0];   // [32,32,32,256] f32
    const float* emb = (const float*)d_in[1];   // [4096,256] f32
    float* out = (float*)d_out;                 // 8388608 z_q floats + 1 loss

    char* ws = (char*)d_ws;
    float*    norms    = (float*)ws;                                   // 16 KB
    _Float16* eh       = (_Float16*)(ws + 16384);                      // 2 MB
    float*    partials = (float*)(ws + 16384 +
                                  (size_t)N_EMB * EMB_DIM * sizeof(_Float16)); // 4 KB

    vq_prep<<<N_EMB, 32, 0, stream>>>(emb, eh, norms);
    vq_main<<<NUM_BLOCKS, WAVES_PER_BLOCK * 32, 0, stream>>>(z, emb, eh, norms, out, partials);
    vq_finalize<<<1, 1, 0, stream>>>(partials, out);
}
